// MuliHeadedMaskedSelfAttention_50508815401252
// MI455X (gfx1250) — matrix operand order repaired and checked by reading, not compiled, not run on hardware
//
#include <hip/hip_runtime.h>
#include <hip/hip_bf16.h>
#include <cstdint>

// ---------------------------------------------------------------------------
// CDNA5 (gfx1250) multi-head causal self-attention, bf16 WMMA pipeline.
//   B=2, T=2048, C=1024, H=16, DK=64
// Compute-bound (~100 GFLOP vs ~60MB HBM) -> v_wmma_f32_16x16x32_bf16 path.
// Tile staging: TDM (tensor_load_to_lds, TENSORcnt) when available, else
// GLOBAL_LOAD_ASYNC_TO_LDS_B128 (ASYNCcnt), else plain LDS stores.
// ---------------------------------------------------------------------------

typedef __attribute__((ext_vector_type(16))) __bf16        v16bf;
typedef __attribute__((ext_vector_type(8)))  float         v8f;
typedef __attribute__((ext_vector_type(4)))  int           v4i;
typedef __attribute__((ext_vector_type(8)))  int           v8i;
typedef __attribute__((ext_vector_type(4)))  unsigned int  v4u;

union Frag {
  v16bf v;
  unsigned int u[8];
};

#ifndef __has_builtin
#define __has_builtin(x) 0
#endif

// ---- CDNA5 async global->LDS copy (GLOBAL_LOAD_ASYNC_TO_LDS_B128) ---------
#if defined(__AMDGCN__) &&                                           \
    __has_builtin(__builtin_amdgcn_global_load_async_to_lds_b128) && \
    __has_builtin(__builtin_amdgcn_s_wait_asynccnt)
#define USE_ASYNC_LDS 1
#else
#define USE_ASYNC_LDS 0
#endif

// ---- CDNA5 Tensor Data Mover (TENSOR_LOAD_TO_LDS) -------------------------
#if defined(__AMDGCN__) &&                                \
    __has_builtin(__builtin_amdgcn_tensor_load_to_lds) && \
    __has_builtin(__builtin_amdgcn_s_wait_tensorcnt)
#define USE_TDM 1
#else
#define USE_TDM 0
#endif

#if USE_ASYNC_LDS
typedef __attribute__((address_space(1))) v4i* as1_v4i;
typedef __attribute__((address_space(3))) v4i* as3_v4i;
#endif
typedef __attribute__((address_space(3))) unsigned short* as3_u16;

__device__ __forceinline__ unsigned lds_offset_of(const void* p) {
  return (unsigned)(unsigned long long)(as3_u16)p;   // LDS byte offset
}

__device__ __forceinline__ void cp16_to_lds(void* lds, const void* g) {
#if USE_ASYNC_LDS
  __builtin_amdgcn_global_load_async_to_lds_b128(
      (as1_v4i)(g), (as3_v4i)(lds), /*imm offset=*/0, /*cpol=*/0);
#else
  *(uint4*)lds = *(const uint4*)g;
#endif
}

__device__ __forceinline__ void async_fence() {
#if USE_ASYNC_LDS
  __builtin_amdgcn_s_wait_asynccnt(0);   // s_wait_asynccnt 0
#endif
}

#if USE_TDM
// 2D tile DMA: global (row-major, elem=2B) -> LDS with per-row padding.
//   tile_w elems per row, tile_h rows, global row stride `stride` elems,
//   LDS padding: after (pad_int code) dwords insert (pad_amt code) dwords.
__device__ __forceinline__ void tdm_load_2d(unsigned lds_off,
                                            unsigned long long gaddr,
                                            unsigned tile_w, unsigned tile_h,
                                            unsigned stride,
                                            unsigned pad_int, unsigned pad_amt) {
  const unsigned dim0 = stride;           // tensor width  (>= tile_w)
  const unsigned dim1 = 1u << 20;         // tensor height (ample; in-bounds use)
  v4u g0;
  g0[0] = 1u;                                             // count=1, user mode
  g0[1] = lds_off;                                        // lds_addr [63:32]
  g0[2] = (unsigned)(gaddr & 0xFFFFFFFFull);              // global_addr lo
  g0[3] = (unsigned)((gaddr >> 32) & 0x01FFFFFFull)       // global_addr[56:32]
          | (2u << 30);                                   // type = 2 (image)
  v8i g1;
  g1[0] = (int)((1u << 16)              // data_size = 1 -> 2 bytes
                | (1u << 20)            // pad_enable
                | (pad_int << 22)       // pad_interval code
                | (pad_amt << 25));     // pad_amount code
  g1[1] = (int)((dim0 & 0xFFFFu) << 16);                  // barrier_addr=0 | dim0 lo
  g1[2] = (int)(((dim0 >> 16) & 0xFFFFu) | ((dim1 & 0xFFFFu) << 16));
  g1[3] = (int)(((dim1 >> 16) & 0xFFFFu) | ((tile_w & 0xFFFFu) << 16));
  g1[4] = (int)(tile_h & 0xFFFFu);                        // tile_dim1 | tile_dim2=0
  g1[5] = (int)stride;                                    // tensor_dim0_stride lo
  g1[6] = 0;                                              // stride hi | dim1_stride lo
  g1[7] = 0;
  const v4i zero4 = {0, 0, 0, 0};                         // groups 2/3: 2D tensor
  const v8i zero8 = {0, 0, 0, 0, 0, 0, 0, 0};
  // 6-arg toolchain flavor: (g0, g1, g2, g3, extra, cpol)
  __builtin_amdgcn_tensor_load_to_lds(g0, g1, zero4, zero4, zero8, /*cpol=*/0);
}
#endif

__device__ __forceinline__ unsigned short f2bf(float f) {
  unsigned int x = __float_as_uint(f);
  unsigned int r = x + 0x7FFFu + ((x >> 16) & 1u);   // round-to-nearest-even
  return (unsigned short)(r >> 16);
}

__device__ __forceinline__ v8f wmma_bf16(const Frag& a, const Frag& b, v8f c) {
  // v_wmma_f32_16x16x32_bf16 : D = A(16x32) * B(32x16) + C(16x16 f32)
  return __builtin_amdgcn_wmma_f32_16x16x32_bf16(
      /*neg_a=*/false, a.v, /*neg_b=*/false, b.v,
      /*c_mod=*/(short)0, c, /*reuse_a=*/false, /*reuse_b=*/false);
}

// ---------------------------------------------------------------------------
// fp32 -> bf16 conversion
// ---------------------------------------------------------------------------
__global__ void f32_to_bf16_kernel(const float* __restrict__ in,
                                   unsigned short* __restrict__ out, int n) {
  int i = blockIdx.x * blockDim.x + threadIdx.x;
  int stride = gridDim.x * blockDim.x;
  for (; i < n; i += stride) out[i] = f2bf(in[i]);
}

// ---------------------------------------------------------------------------
// Tiled bf16 WMMA GEMM:  C[M,N] = A[M,K] * B[K,N], row-major, f32 accumulate.
// Block = 256 threads (8 waves), tile 128x128, K-step 32, double-buffered LDS.
// A tile staged by TDM (wave 0 posts descriptor) or async copies.
// Waves arranged 4(M) x 2(N); each wave computes 32x64 via 2x4 WMMA tiles.
// ---------------------------------------------------------------------------
template <int WRITE_F32>
__global__ __launch_bounds__(256)
void gemm_bf16_kernel(const unsigned short* __restrict__ A,
                      const unsigned short* __restrict__ Bm,
                      float* __restrict__ Cf,
                      unsigned short* __restrict__ Cb,
                      int M, int N, int K) {
  constexpr int BM = 128, BN = 128, BK = 32;
  constexpr int LDA = 40;  // padded row stride (elems): conflict-free, 16B aligned
  constexpr int LDB = 40;
  __shared__ unsigned short sA[2][BM * LDA];   // [m][k], double buffered
  __shared__ unsigned short sB[2][BN * LDB];   // [n][k]  (B transposed)

  const int tid  = threadIdx.x;
  const int lane = tid & 31;
  const int wid  = tid >> 5;
  const int wm   = wid & 3;        // wave position along M (4)
  const int wn   = wid >> 2;       // wave position along N (2)
  const int half = lane >> 4;      // 0: lanes 0-15, 1: lanes 16-31
  const int l16  = lane & 15;
  const int bm0  = blockIdx.y * BM;
  const int bn0  = blockIdx.x * BN;

  const v8f vzero = {0.f, 0.f, 0.f, 0.f, 0.f, 0.f, 0.f, 0.f};
  v8f acc[2][4];
#pragma unroll
  for (int i = 0; i < 2; ++i)
#pragma unroll
    for (int j = 0; j < 4; ++j) acc[i][j] = vzero;

  // ---- staging helpers ----------------------------------------------------
  auto stageA = [&](int buf, int kt) {
#if USE_TDM
    if (wid == 0) {
      // 128 rows x 32 elems (64B row = 16 dwords), LDS row stride 40 elems:
      // pad_interval code 3 (16 dwords), pad_amount code 3 (4 dwords).
      tdm_load_2d(lds_offset_of(&sA[buf][0]),
                  (unsigned long long)(size_t)(A + (size_t)bm0 * K + kt),
                  /*tile_w=*/BK, /*tile_h=*/BM, /*stride=*/(unsigned)K,
                  /*pad_int=*/3, /*pad_amt=*/3);
    }
#else
#pragma unroll
    for (int it = 0; it < 2; ++it) {
      int idx = tid + it * 256;            // 0..511
      int row = idx >> 2;                  // 0..127
      int kof = (idx & 3) * 8;             // 0,8,16,24
      cp16_to_lds(&sA[buf][row * LDA + kof],
                  A + (size_t)(bm0 + row) * K + kt + kof);
    }
#endif
  };
  auto stageB = [&](int buf, int kt) {  // transposing copy (synchronous)
    int n  = tid >> 1;                   // 0..127
    int k0 = (tid & 1) * 16;             // 0 or 16
    const unsigned short* g = Bm + (size_t)(kt + k0) * N + (bn0 + n);
    unsigned int d[8];
#pragma unroll
    for (int j = 0; j < 8; ++j) {
      unsigned int lo = g[(size_t)(2 * j) * N];
      unsigned int hi = g[(size_t)(2 * j + 1) * N];
      d[j] = lo | (hi << 16);
    }
    uint4* p = (uint4*)(&sB[buf][n * LDB + k0]);
    p[0] = make_uint4(d[0], d[1], d[2], d[3]);
    p[1] = make_uint4(d[4], d[5], d[6], d[7]);
  };

  const int nkt = K / BK;
  stageA(0, 0);
  stageB(0, 0);

  for (int ki = 0; ki < nkt; ++ki) {
    const int cur = ki & 1;
#if USE_TDM
    if (wid == 0) __builtin_amdgcn_s_wait_tensorcnt(0);
#else
    async_fence();     // my async copies for buffer `cur` have landed
#endif
    __syncthreads();   // everyone's staging for `cur` visible
    if (ki + 1 < nkt) {  // prefetch next tile into the other buffer
      stageA(cur ^ 1, (ki + 1) * BK);
      stageB(cur ^ 1, (ki + 1) * BK);
    }

    // ---- load fragments from LDS per ISA 7.12.2 layouts -------------------
    Frag af[2], bfr[4];
#pragma unroll
    for (int mf = 0; mf < 2; ++mf) {
      int m = wm * 32 + mf * 16 + l16;
#pragma unroll
      for (int vv = 0; vv < 8; ++vv) {
        int k = (vv >> 2) * 16 + half * 8 + (vv & 3) * 2;   // A 16x32 map
        af[mf].u[vv] = *(const unsigned int*)(&sA[cur][m * LDA + k]);
      }
    }
#pragma unroll
    for (int nf = 0; nf < 4; ++nf) {
      int n = wn * 64 + nf * 16 + l16;
#pragma unroll
      for (int vv = 0; vv < 8; ++vv) {
        int k = (vv >> 2) * 16 + half * 8 + (vv & 3) * 2;   // B 32x16 map
        bfr[nf].u[vv] = *(const unsigned int*)(&sB[cur][n * LDB + k]);
      }
    }
#pragma unroll
    for (int mf = 0; mf < 2; ++mf)
#pragma unroll
      for (int nf = 0; nf < 4; ++nf)
        acc[mf][nf] = wmma_bf16(af[mf], bfr[nf], acc[mf][nf]);
  }

  // ---- store (C layout: VGPR r, lane l -> row r+8*half, col l16) ----------
#pragma unroll
  for (int mf = 0; mf < 2; ++mf)
#pragma unroll
    for (int nf = 0; nf < 4; ++nf)
#pragma unroll
      for (int r = 0; r < 8; ++r) {
        int row = bm0 + wm * 32 + mf * 16 + r + 8 * half;
        int col = bn0 + wn * 64 + nf * 16 + l16;
        float val = acc[mf][nf][r];
        if (WRITE_F32)
          Cf[(size_t)row * N + col] = val;
        else
          Cb[(size_t)row * N + col] = f2bf(val);
      }
}

// ---------------------------------------------------------------------------
// Flash attention (causal, scores *multiplied* by sqrt(d)=8 per reference).
// Grid: (T/64, B*H). Block: 128 threads = 4 waves, each wave owns 16 queries.
// q/k/v layout: [B, T, H, DK] bf16.  Output o: [B, T, H*DK] bf16.
// K chunk staged by TDM; V chunk transposed synchronously.
// ---------------------------------------------------------------------------
__global__ __launch_bounds__(128)
void attn_kernel(const unsigned short* __restrict__ qg,
                 const unsigned short* __restrict__ kg,
                 const unsigned short* __restrict__ vg,
                 unsigned short* __restrict__ og) {
  constexpr int T = 2048, H = 16, DK = 64;
  constexpr int LDK = 72;  // sK row stride (elems)
  constexpr int LDV = 40;  // sV row stride (elems)
  constexpr int LDP = 40;  // sP row stride (elems)
  __shared__ unsigned short sK[32 * LDK];        // [key][d]
  __shared__ unsigned short sV[64 * LDV];        // [d][key]   (transposed)
  __shared__ unsigned short sP[4 * 16 * LDP];    // per-wave P (16x32)

  const int tid  = threadIdx.x;
  const int lane = tid & 31;
  const int wid  = tid >> 5;
  const int half = lane >> 4;
  const int l16  = lane & 15;
  const int b    = blockIdx.y >> 4;
  const int h    = blockIdx.y & 15;
  const int qbase = blockIdx.x * 64;
  const int qw    = qbase + wid * 16;            // wave's first query row

  // ---- preload Q strip (16x64) as two A fragments -------------------------
  Frag qa[2];
  {
    const unsigned short* qrow =
        qg + ((size_t)(b * T + qw + l16) * H + h) * DK;
#pragma unroll
    for (int f = 0; f < 2; ++f)
#pragma unroll
      for (int vv = 0; vv < 8; ++vv) {
        int d = f * 32 + (vv >> 2) * 16 + half * 8 + (vv & 3) * 2;
        qa[f].u[vv] = *(const unsigned int*)(qrow + d);
      }
  }

  const v8f vzero = {0.f, 0.f, 0.f, 0.f, 0.f, 0.f, 0.f, 0.f};
  v8f oacc[4];
#pragma unroll
  for (int i = 0; i < 4; ++i) oacc[i] = vzero;
  float rmax[8], rsum[8];
#pragma unroll
  for (int r = 0; r < 8; ++r) { rmax[r] = -1e30f; rsum[r] = 0.f; }

  const int nchunk = qbase / 32 + 2;             // causal: keys <= qbase+63
  for (int c = 0; c < nchunk; ++c) {
    const int kb = c * 32;
    __syncthreads();                             // protect LDS reuse
    // ---- stage K chunk (32 keys x 64 d) -----------------------------------
#if USE_TDM
    if (wid == 0) {
      // 32 rows x 64 elems (128B row = 32 dwords), LDS row stride 72 elems:
      // pad_interval code 4 (32 dwords), pad_amount code 3 (4 dwords).
      tdm_load_2d(lds_offset_of(&sK[0]),
                  (unsigned long long)(size_t)(
                      kg + ((size_t)(b * T + kb) * H + h) * DK),
                  /*tile_w=*/DK, /*tile_h=*/32, /*stride=*/H * DK,
                  /*pad_int=*/4, /*pad_amt=*/3);
    }
#else
#pragma unroll
    for (int it = 0; it < 2; ++it) {
      int idx = tid + it * 128;                  // 0..255
      int key = idx >> 3;                        // 0..31
      int dof = (idx & 7) * 8;                   // 0..56
      cp16_to_lds(&sK[key * LDK + dof],
                  kg + ((size_t)(b * T + kb + key) * H + h) * DK + dof);
    }
#endif
    // ---- stage V chunk transposed: sV[d][key] (synchronous) ---------------
    {
      int d  = tid >> 1;                         // 0..63
      int k0 = (tid & 1) * 16;                   // 0 or 16
      const unsigned short* g =
          vg + ((size_t)(b * T + kb + k0) * H + h) * DK + d;
      unsigned int w[8];
#pragma unroll
      for (int j = 0; j < 8; ++j) {
        unsigned int lo = g[(size_t)(2 * j) * H * DK];
        unsigned int hi = g[(size_t)(2 * j + 1) * H * DK];
        w[j] = lo | (hi << 16);
      }
      uint4* p = (uint4*)(&sV[d * LDV + k0]);
      p[0] = make_uint4(w[0], w[1], w[2], w[3]);
      p[1] = make_uint4(w[4], w[5], w[6], w[7]);
    }
#if USE_TDM
    if (wid == 0) __builtin_amdgcn_s_wait_tensorcnt(0);
#else
    async_fence();
#endif
    __syncthreads();

    const bool active = (kb <= qw + 15);         // wave-uniform causal guard
    if (active) {
      // ---- S = Q * K^T : two 16x16 tiles over 32 keys ---------------------
      v8f s[2];
#pragma unroll
      for (int nt = 0; nt < 2; ++nt) {
        v8f t = vzero;
#pragma unroll
        for (int f = 0; f < 2; ++f) {
          Frag kbf;
#pragma unroll
          for (int vv = 0; vv < 8; ++vv) {
            int d   = f * 32 + (vv >> 2) * 16 + half * 8 + (vv & 3) * 2;
            int key = nt * 16 + l16;
            kbf.u[vv] = *(const unsigned int*)(&sK[key * LDK + d]);
          }
          t = wmma_bf16(qa[f], kbf, t);
        }
        s[nt] = t;
      }
      // ---- scale (*sqrt(64)=8), causal mask, online softmax ---------------
      unsigned short* pw = &sP[wid * 16 * LDP];
#pragma unroll
      for (int r = 0; r < 8; ++r) {
        const int qrow = qw + r + 8 * half;
        float a0 = s[0][r] * 8.0f;
        float a1 = s[1][r] * 8.0f;
        if (kb + l16 > qrow)      a0 = -__builtin_inff();
        if (kb + 16 + l16 > qrow) a1 = -__builtin_inff();
        float mx = fmaxf(a0, a1);
        mx = fmaxf(mx, __shfl_xor(mx, 1, 32));
        mx = fmaxf(mx, __shfl_xor(mx, 2, 32));
        mx = fmaxf(mx, __shfl_xor(mx, 4, 32));
        mx = fmaxf(mx, __shfl_xor(mx, 8, 32));
        const float nm    = fmaxf(rmax[r], mx);
        const float alpha = __expf(rmax[r] - nm);
        const float p0 = __expf(a0 - nm);
        const float p1 = __expf(a1 - nm);
        float sum = p0 + p1;
        sum += __shfl_xor(sum, 1, 32);
        sum += __shfl_xor(sum, 2, 32);
        sum += __shfl_xor(sum, 4, 32);
        sum += __shfl_xor(sum, 8, 32);
        rsum[r] = rsum[r] * alpha + sum;
        rmax[r] = nm;
#pragma unroll
        for (int nt = 0; nt < 4; ++nt) oacc[nt][r] *= alpha;
        // C-layout (row r+8*half, col l16) -> row-major P block in LDS
        const int m = r + 8 * half;
        pw[m * LDP + l16]      = f2bf(p0);
        pw[m * LDP + 16 + l16] = f2bf(p1);
      }
    }
    __syncthreads();                             // P visible (also syncs block)
    if (active) {
      // ---- O += P * V : P (16x32) as A-fragment, V chunk as B -------------
      Frag pa;
#pragma unroll
      for (int vv = 0; vv < 8; ++vv) {
        int n = (vv >> 2) * 16 + half * 8 + (vv & 3) * 2;
        pa.u[vv] =
            *(const unsigned int*)(&sP[wid * 16 * LDP + l16 * LDP + n]);
      }
#pragma unroll
      for (int nt = 0; nt < 4; ++nt) {
        Frag vb;
#pragma unroll
        for (int vv = 0; vv < 8; ++vv) {
          int key = (vv >> 2) * 16 + half * 8 + (vv & 3) * 2;
          int d   = nt * 16 + l16;
          vb.u[vv] = *(const unsigned int*)(&sV[d * LDV + key]);
        }
        oacc[nt] = wmma_bf16(pa, vb, oacc[nt]);
      }
    }
  }

  // ---- normalize and store merged-head o: [B, T, H*DK] bf16 ---------------
#pragma unroll
  for (int nt = 0; nt < 4; ++nt)
#pragma unroll
    for (int r = 0; r < 8; ++r) {
      const int m    = r + 8 * half;
      const int trow = qw + m;
      const int d    = nt * 16 + l16;
      const float val = oacc[nt][r] / rsum[r];
      og[(size_t)(b * T + trow) * (H * DK) + h * DK + d] = f2bf(val);
    }
}

// ---------------------------------------------------------------------------
// Host-side launcher
// ---------------------------------------------------------------------------
extern "C" void kernel_launch(void* const* d_in, const int* in_sizes, int n_in,
                              void* d_out, int out_size, void* d_ws,
                              size_t ws_size, hipStream_t stream) {
  (void)in_sizes; (void)n_in; (void)out_size; (void)ws_size;
  constexpr int Bb = 2, T = 2048, C = 1024, H = 16;
  const size_t NX = (size_t)Bb * T * C;   // 4,194,304 activations
  const size_t NW = (size_t)C * C;        // 1,048,576 weight elems

  const float* x  = (const float*)d_in[0];
  const float* Qw = (const float*)d_in[1];
  const float* Kw = (const float*)d_in[2];
  const float* Vw = (const float*)d_in[3];
  const float* Ow = (const float*)d_in[4];
  float* out = (float*)d_out;

  // workspace carve-out (~48 MB), 256B aligned slabs
  char* ws = (char*)d_ws;
  size_t off = 0;
  auto carve = [&](size_t elems) {
    unsigned short* p = (unsigned short*)(ws + off);
    off = (off + elems * 2 + 255) & ~(size_t)255;
    return p;
  };
  unsigned short* xb = carve(NX);
  unsigned short* Qb = carve(NW);
  unsigned short* Kb = carve(NW);
  unsigned short* Vb = carve(NW);
  unsigned short* Ob = carve(NW);
  unsigned short* qb = carve(NX);
  unsigned short* kb = carve(NX);
  unsigned short* vb = carve(NX);
  unsigned short* ob = carve(NX);

  // 1) convert fp32 -> bf16
  f32_to_bf16_kernel<<<512, 256, 0, stream>>>(x,  xb, (int)NX);
  f32_to_bf16_kernel<<<256, 256, 0, stream>>>(Qw, Qb, (int)NW);
  f32_to_bf16_kernel<<<256, 256, 0, stream>>>(Kw, Kb, (int)NW);
  f32_to_bf16_kernel<<<256, 256, 0, stream>>>(Vw, Vb, (int)NW);
  f32_to_bf16_kernel<<<256, 256, 0, stream>>>(Ow, Ob, (int)NW);

  // 2) q/k/v projections: [4096,1024] x [1024,1024] -> bf16 [B,T,H,DK]
  dim3 gB(256), gG(C / 128, (Bb * T) / 128);
  gemm_bf16_kernel<0><<<gG, gB, 0, stream>>>(xb, Qb, nullptr, qb, Bb * T, C, C);
  gemm_bf16_kernel<0><<<gG, gB, 0, stream>>>(xb, Kb, nullptr, kb, Bb * T, C, C);
  gemm_bf16_kernel<0><<<gG, gB, 0, stream>>>(xb, Vb, nullptr, vb, Bb * T, C, C);

  // 3) causal flash attention
  dim3 aB(128), aG(T / 64, Bb * H);
  attn_kernel<<<aG, aB, 0, stream>>>(qb, kb, vb, ob);

  // 4) output projection -> fp32 d_out
  gemm_bf16_kernel<1><<<gG, gB, 0, stream>>>(ob, Ob, out, nullptr, Bb * T, C, C);
}
